// CausalWanSelfAttention_8615704395907
// MI455X (gfx1250) — matrix operand order repaired and checked by reading, not compiled, not run on hardware
//
#include <hip/hip_runtime.h>
#include <hip/hip_bf16.h>

// ---------------------------------------------------------------------------
// CausalWanSelfAttention for MI455X (gfx1250).
// bf16 WMMA (v_wmma_f32_16x16x32_bf16) everywhere, fp32 accumulate.
// GEMM is register-blocked 32x64 per wave (8 WMMAs per k-step, 12 b128 loads).
// Workspace use: ~128 MB of d_ws.
// ---------------------------------------------------------------------------

typedef __attribute__((ext_vector_type(16))) __bf16 v16bf;
typedef __attribute__((ext_vector_type(8)))  __bf16 v8bf;
typedef __attribute__((ext_vector_type(8)))  float  v8f;

#define WMMA_BF16(a, b, c) \
  __builtin_amdgcn_wmma_f32_16x16x32_bf16(false, (a), false, (b), (short)0, (c), false, false)

namespace {
constexpr int S    = 1760;   // new tokens
constexpr int DIM  = 2048;
constexpr int NH   = 16;
constexpr int HD   = 128;    // head dim
constexpr int KW   = 3520;   // attention key window (= MAX_ATT)
constexpr int COFF = 2640;   // cache_k[COFF + j] feeds window slot j (j < 1760)
constexpr float ATT_SCALE = 0.08838834764831845f; // 1/sqrt(128)
}

// -- fragment load helpers ---------------------------------------------------
__device__ inline v16bf ld2(const __bf16* p0, const __bf16* p1) {
  union { v16bf v; v8bf h[2]; } u;
  u.h[0] = *(const v8bf*)p0;
  u.h[1] = *(const v8bf*)p1;
  return u.v;
}
// A-matrix 16x32 bf16 fragment: lane(row r, half hf) holds K = hf*8+{0..7} and 16+hf*8+{0..7}
__device__ inline v16bf load_a(const __bf16* rowbase_k0, int hf) {
  return ld2(rowbase_k0 + hf * 8, rowbase_k0 + hf * 8 + 16);
}
// B-matrix 32x16 bf16 fragment: lane(col r, half hf) holds K = hf*8+{0..7} and 16+hf*8+{0..7}
__device__ inline v16bf load_b(const __bf16* colbase_k0, int hf) {
  return ld2(colbase_k0 + hf * 8, colbase_k0 + hf * 8 + 16);
}

// -- fp32 -> bf16 convert ----------------------------------------------------
__global__ void cvt_bf16_kernel(const float* __restrict__ src,
                                __bf16* __restrict__ dst, size_t n) {
  size_t i = (size_t)blockIdx.x * blockDim.x + threadIdx.x;
  size_t stride = (size_t)gridDim.x * blockDim.x;
  for (; i < n; i += stride) dst[i] = (__bf16)src[i];
}

// -- GEMM: out[M][N] = A[M][K] * Bw[N][K]^T + bias[N]  (bf16 in, f32 out) ----
// 1 wave -> 32x64 tile (2 A-frags x 4 B-frags, 8 WMMAs / k-step).
// Block = 128 threads = 4 waves along N -> block tile 32x256.
__global__ __launch_bounds__(128)
void gemm_bf16_kernel(const __bf16* __restrict__ A, const __bf16* __restrict__ Bw,
                      const float* __restrict__ bias, float* __restrict__ out,
                      int M, int N, int Kd) {
  const int lane = threadIdx.x & 31;
  const int wave = threadIdx.x >> 5;
  const int r  = lane & 15;
  const int hf = lane >> 4;
  const int n0 = blockIdx.x * 256 + wave * 64;   // 4 N-tiles of 16
  const int m0 = blockIdx.y * 32;                // 2 M-tiles of 16

  const __bf16* ap0 = A + (size_t)(m0 + r) * Kd;
  const __bf16* ap1 = A + (size_t)(m0 + 16 + r) * Kd;
  const __bf16* bp0 = Bw + (size_t)(n0 + r) * Kd;
  const __bf16* bp1 = Bw + (size_t)(n0 + 16 + r) * Kd;
  const __bf16* bp2 = Bw + (size_t)(n0 + 32 + r) * Kd;
  const __bf16* bp3 = Bw + (size_t)(n0 + 48 + r) * Kd;

  const v8f vz = {0.f, 0.f, 0.f, 0.f, 0.f, 0.f, 0.f, 0.f};
  v8f acc[2][4];
#pragma unroll
  for (int t = 0; t < 2; ++t)
#pragma unroll
    for (int j = 0; j < 4; ++j) acc[t][j] = vz;

#pragma unroll 2
  for (int k = 0; k < Kd; k += 32) {
    const v16bf a0 = load_a(ap0 + k, hf);
    const v16bf a1 = load_a(ap1 + k, hf);
    {
      const v16bf b0 = load_b(bp0 + k, hf);
      acc[0][0] = WMMA_BF16(a0, b0, acc[0][0]);
      acc[1][0] = WMMA_BF16(a1, b0, acc[1][0]);
    }
    {
      const v16bf b1 = load_b(bp1 + k, hf);
      acc[0][1] = WMMA_BF16(a0, b1, acc[0][1]);
      acc[1][1] = WMMA_BF16(a1, b1, acc[1][1]);
    }
    {
      const v16bf b2 = load_b(bp2 + k, hf);
      acc[0][2] = WMMA_BF16(a0, b2, acc[0][2]);
      acc[1][2] = WMMA_BF16(a1, b2, acc[1][2]);
    }
    {
      const v16bf b3 = load_b(bp3 + k, hf);
      acc[0][3] = WMMA_BF16(a0, b3, acc[0][3]);
      acc[1][3] = WMMA_BF16(a1, b3, acc[1][3]);
    }
  }

#pragma unroll
  for (int j = 0; j < 4; ++j) {
    const float bb = bias[n0 + j * 16 + r];
#pragma unroll
    for (int t = 0; t < 2; ++t) {
#pragma unroll
      for (int i = 0; i < 8; ++i)
        out[(size_t)(m0 + t * 16 + i + 8 * hf) * N + (n0 + j * 16 + r)] =
            acc[t][j][i] + bb;
    }
  }
}

// -- rms_norm (over full 2048) + RoPE, write head-major bf16 -----------------
// dst[n * headStride + (sOff + s) * HD + d]; postScale folds 1/sqrt(HD) for Q.
__global__ __launch_bounds__(256)
void normrope_kernel(const float* __restrict__ y, const float* __restrict__ g,
                     const float* __restrict__ freqs, __bf16* __restrict__ dst,
                     int headStride, int sOff, float postScale) {
  const int s = blockIdx.x;
  const float* row = y + (size_t)s * DIM;
  __shared__ float red[8];

  float ss = 0.f;
  for (int i = threadIdx.x; i < DIM; i += 256) { float v = row[i]; ss += v * v; }
#pragma unroll
  for (int msk = 1; msk < 32; msk <<= 1) ss += __shfl_xor(ss, msk, 32);
  if ((threadIdx.x & 31) == 0) red[threadIdx.x >> 5] = ss;
  __syncthreads();
  float tot = 0.f;
#pragma unroll
  for (int w = 0; w < 8; ++w) tot += red[w];
  const float rms = rsqrtf(tot / (float)DIM + 1e-6f) * postScale;

  // token -> (frame, h, w);  START_FRAME = 5, H*W = 880, W = 44
  const int f = s / 880, rem = s % 880, hh = rem / 44, ww = rem % 44;

  for (int p = threadIdx.x; p < DIM / 2; p += 256) {
    const int n = p >> 6;          // head
    const int c = p & 63;          // complex pair within head
    const int frow = (c < 22) ? (5 + f) : ((c < 43) ? hh : ww);
    const float fr = freqs[((size_t)frow * 64 + c) * 2 + 0];
    const float fi = freqs[((size_t)frow * 64 + c) * 2 + 1];
    const float xr = row[2 * p]     * rms * g[2 * p];
    const float xi = row[2 * p + 1] * rms * g[2 * p + 1];
    __bf16* d = dst + (size_t)n * headStride + (size_t)(sOff + s) * HD + 2 * c;
    d[0] = (__bf16)(xr * fr - xi * fi);
    d[1] = (__bf16)(xr * fi + xi * fr);
  }
}

// -- window builders ---------------------------------------------------------
// kwin[n][j][d] = cache_k[COFF + j][n][d]   (j < 1760; rolled region of cache)
__global__ void gather_cache_k_kernel(const float* __restrict__ ck,
                                      __bf16* __restrict__ kwin) {
  const size_t total = (size_t)NH * S * HD;
  size_t i = (size_t)blockIdx.x * blockDim.x + threadIdx.x;
  size_t stride = (size_t)gridDim.x * blockDim.x;
  for (; i < total; i += stride) {
    const int d = (int)(i & (HD - 1));
    const size_t t = i >> 7;
    const int j = (int)(t % S);
    const int n = (int)(t / S);
    kwin[((size_t)n * KW + j) * HD + d] =
        (__bf16)ck[((size_t)(COFF + j) * NH + n) * HD + d];
  }
}
// vwt[n][d][j] = cache_v[COFF + j][n][d]   (transposed, d-major)
__global__ void gather_cache_v_kernel(const float* __restrict__ cv,
                                      __bf16* __restrict__ vwt) {
  const size_t total = (size_t)NH * S * HD;
  size_t i = (size_t)blockIdx.x * blockDim.x + threadIdx.x;
  size_t stride = (size_t)gridDim.x * blockDim.x;
  for (; i < total; i += stride) {
    const int d = (int)(i & (HD - 1));
    const size_t t = i >> 7;
    const int j = (int)(t % S);
    const int n = (int)(t / S);
    vwt[((size_t)n * HD + d) * KW + j] =
        (__bf16)cv[((size_t)(COFF + j) * NH + n) * HD + d];
  }
}
// vwt[n][d][1760 + s] = yv[s][n*128 + d]
__global__ void scatter_v_kernel(const float* __restrict__ yv,
                                 __bf16* __restrict__ vwt) {
  const size_t total = (size_t)S * DIM;
  size_t i = (size_t)blockIdx.x * blockDim.x + threadIdx.x;
  size_t stride = (size_t)gridDim.x * blockDim.x;
  for (; i < total; i += stride) {
    const int col = (int)(i & (DIM - 1));
    const size_t s = i >> 11;
    const int n = col >> 7, d = col & 127;
    vwt[((size_t)n * HD + d) * KW + (S + s)] = (__bf16)yv[i];
  }
}

// -- flash attention: 1 wave per (head, 16-query tile) -----------------------
__global__ __launch_bounds__(32)
void attn_kernel(const __bf16* __restrict__ rq, const __bf16* __restrict__ kwin,
                 const __bf16* __restrict__ vwt, __bf16* __restrict__ ob) {
  __shared__ __bf16 pstage[16 * 32];   // P tile, D-layout -> A-layout staging

  const int lane = threadIdx.x;
  const int r  = lane & 15;
  const int hf = lane >> 4;
  const int head = blockIdx.y;
  const int q0   = blockIdx.x * 16;

  const __bf16* Q  = rq   + ((size_t)head * S + q0) * HD;  // [16][128]
  const __bf16* K  = kwin + (size_t)head * KW * HD;        // [KW][128]
  const __bf16* Vt = vwt  + (size_t)head * HD * KW;        // [128][KW]

  // Q A-fragments for the 4 K-dim chunks of 32 (scale pre-folded into rq)
  v16bf qf[4];
#pragma unroll
  for (int c = 0; c < 4; ++c) qf[c] = load_a(Q + (size_t)r * HD + c * 32, hf);

  v8f o[8];
  const v8f vz = {0.f, 0.f, 0.f, 0.f, 0.f, 0.f, 0.f, 0.f};
#pragma unroll
  for (int dt = 0; dt < 8; ++dt) o[dt] = vz;
  float mrow[8], lrow[8];
#pragma unroll
  for (int i = 0; i < 8; ++i) { mrow[i] = -3.0e38f; lrow[i] = 0.f; }

  for (int kb = 0; kb < KW; kb += 32) {
    __builtin_prefetch(K + (size_t)(kb + 32 + r) * HD, 0, 1);
    __builtin_prefetch(Vt + (size_t)r * KW + kb + 32, 0, 1);

    // ---- S = Q * K^T : two 16x16 score tiles (keys kb..kb+15, kb+16..kb+31)
    v8f s0 = vz, s1 = vz;
#pragma unroll
    for (int c = 0; c < 4; ++c) {
      v16bf k0 = load_b(K + (size_t)(kb + r) * HD + c * 32, hf);
      v16bf k1 = load_b(K + (size_t)(kb + 16 + r) * HD + c * 32, hf);
      s0 = WMMA_BF16(qf[c], k0, s0);
      s1 = WMMA_BF16(qf[c], k1, s1);
    }

    // ---- online softmax (rows live on 16 lanes of one half-wave)
#pragma unroll
    for (int i = 0; i < 8; ++i) {
      float a = s0[i], b = s1[i];
      float mx = fmaxf(a, b);
#pragma unroll
      for (int msk = 1; msk < 16; msk <<= 1) mx = fmaxf(mx, __shfl_xor(mx, msk, 32));
      const float mnew = fmaxf(mrow[i], mx);
      const float corr = __expf(mrow[i] - mnew);
      const float p0 = __expf(a - mnew);
      const float p1 = __expf(b - mnew);
      float rs = p0 + p1;
#pragma unroll
      for (int msk = 1; msk < 16; msk <<= 1) rs += __shfl_xor(rs, msk, 32);
      lrow[i] = lrow[i] * corr + rs;
      mrow[i] = mnew;
#pragma unroll
      for (int dt = 0; dt < 8; ++dt) o[dt][i] *= corr;
      const int row = i + 8 * hf;
      pstage[row * 32 + r]      = (__bf16)p0;
      pstage[row * 32 + 16 + r] = (__bf16)p1;
    }
    // per-wave LDS is in-order, but make the D->A relayout explicit
    asm volatile("s_wait_dscnt 0" ::: "memory");

    // ---- P A-fragment (16x32) from LDS
    v16bf pf = ld2((const __bf16*)&pstage[r * 32 + hf * 8],
                   (const __bf16*)&pstage[r * 32 + 16 + hf * 8]);

    // ---- O += P * V : 8 d-tiles, V B-fragments contiguous from vwt
#pragma unroll
    for (int dt = 0; dt < 8; ++dt) {
      v16bf vf = load_b(Vt + (size_t)(dt * 16 + r) * KW + kb, hf);
      o[dt] = WMMA_BF16(pf, vf, o[dt]);
    }
  }

  // ---- epilogue: O /= l, write bf16 row-major [S][DIM] for the final GEMM
#pragma unroll
  for (int i = 0; i < 8; ++i) {
    const float inv = 1.0f / lrow[i];
    const int row = q0 + i + 8 * hf;
#pragma unroll
    for (int dt = 0; dt < 8; ++dt)
      ob[(size_t)row * DIM + head * HD + dt * 16 + r] = (__bf16)(o[dt][i] * inv);
  }
}

// ---------------------------------------------------------------------------
extern "C" void kernel_launch(void* const* d_in, const int* in_sizes, int n_in,
                              void* d_out, int out_size, void* d_ws, size_t ws_size,
                              hipStream_t stream) {
  (void)in_sizes; (void)n_in; (void)out_size; (void)ws_size;
  const float* x   = (const float*)d_in[0];
  const float* wq  = (const float*)d_in[1];
  const float* bq  = (const float*)d_in[2];
  const float* wk  = (const float*)d_in[3];
  const float* bk  = (const float*)d_in[4];
  const float* wv  = (const float*)d_in[5];
  const float* bv  = (const float*)d_in[6];
  const float* wo  = (const float*)d_in[7];
  const float* bo  = (const float*)d_in[8];
  const float* gq  = (const float*)d_in[9];
  const float* gk  = (const float*)d_in[10];
  const float* fre = (const float*)d_in[11];
  const float* ck  = (const float*)d_in[12];
  const float* cv  = (const float*)d_in[13];
  float* out = (float*)d_out;

  char* base = (char*)d_ws;
  size_t off = 0;
  auto alloc = [&](size_t bytes) -> void* {
    off = (off + 255) & ~(size_t)255;
    void* p = base + off;
    off += bytes;
    return p;
  };
  __bf16* XB   = (__bf16*)alloc((size_t)S * DIM * 2);
  __bf16* WQB  = (__bf16*)alloc((size_t)DIM * DIM * 2);
  __bf16* WKB  = (__bf16*)alloc((size_t)DIM * DIM * 2);
  __bf16* WVB  = (__bf16*)alloc((size_t)DIM * DIM * 2);
  __bf16* WOB  = (__bf16*)alloc((size_t)DIM * DIM * 2);
  float*  YQ   = (float*)alloc((size_t)S * DIM * 4);
  float*  YK   = (float*)alloc((size_t)S * DIM * 4);
  float*  YV   = (float*)alloc((size_t)S * DIM * 4);
  __bf16* RQ   = (__bf16*)alloc((size_t)NH * S * HD * 2);
  __bf16* KWIN = (__bf16*)alloc((size_t)NH * KW * HD * 2);
  __bf16* VWT  = (__bf16*)alloc((size_t)NH * HD * KW * 2);
  __bf16* OB   = (__bf16*)alloc((size_t)S * DIM * 2);

  // 1) fp32 -> bf16 converts
  cvt_bf16_kernel<<<2048, 256, 0, stream>>>(x,  XB,  (size_t)S * DIM);
  cvt_bf16_kernel<<<2048, 256, 0, stream>>>(wq, WQB, (size_t)DIM * DIM);
  cvt_bf16_kernel<<<2048, 256, 0, stream>>>(wk, WKB, (size_t)DIM * DIM);
  cvt_bf16_kernel<<<2048, 256, 0, stream>>>(wv, WVB, (size_t)DIM * DIM);
  cvt_bf16_kernel<<<2048, 256, 0, stream>>>(wo, WOB, (size_t)DIM * DIM);

  // 2) QKV projections (WMMA GEMM, bias fused, fp32 out)
  dim3 gg(DIM / 256, S / 32);   // (8, 55): 32x256 block tile, 32x64 per wave
  gemm_bf16_kernel<<<gg, 128, 0, stream>>>(XB, WQB, bq, YQ, S, DIM, DIM);
  gemm_bf16_kernel<<<gg, 128, 0, stream>>>(XB, WKB, bk, YK, S, DIM, DIM);
  gemm_bf16_kernel<<<gg, 128, 0, stream>>>(XB, WVB, bv, YV, S, DIM, DIM);

  // 3) rms_norm + rope; Q gets 1/sqrt(128) folded in
  normrope_kernel<<<S, 256, 0, stream>>>(YQ, gq, fre, RQ,   S  * HD, 0, ATT_SCALE);
  normrope_kernel<<<S, 256, 0, stream>>>(YK, gk, fre, KWIN, KW * HD, S, 1.0f);

  // 4) key/value windows (rolled cache half + new half)
  gather_cache_k_kernel<<<4096, 256, 0, stream>>>(ck, KWIN);
  gather_cache_v_kernel<<<4096, 256, 0, stream>>>(cv, VWT);
  scatter_v_kernel<<<4096, 256, 0, stream>>>(YV, VWT);

  // 5) flash attention (WMMA QK^T + PV)
  attn_kernel<<<dim3(S / 16, NH), 32, 0, stream>>>(RQ, KWIN, VWT, OB);

  // 6) output projection -> d_out (fp32)
  gemm_bf16_kernel<<<gg, 128, 0, stream>>>(OB, WOB, bo, out, S, DIM, DIM);
}
